// TransformerRadford_48077863911615
// MI455X (gfx1250) — matrix operand order repaired and checked by reading, not compiled, not run on hardware
//
#include <hip/hip_runtime.h>
#include <hip/hip_bf16.h>

// ---------------------------------------------------------------------------
// Transformer block (attention + FFN + 2x LayerNorm) for MI455X / gfx1250.
// GEMMs: v_wmma_f32_16x16x32_bf16 (bf16 in, f32 accumulate).
// Global->LDS feeding: GLOBAL_LOAD_ASYNC_TO_LDS_B128 (ASYNCcnt-tracked DMA,
// no VGPR staging, no spills), triple-buffered LDS, s_wait_asynccnt + barrier.
// Workspace requirement: ~390 MB (buffers aliased; see kernel_launch).
// Shapes fixed to the reference: B=4, S=2048, E=D=768, H=12, dh=64, F=3072.
// ---------------------------------------------------------------------------

typedef __attribute__((ext_vector_type(16))) __bf16 v16bf;
typedef __attribute__((ext_vector_type(8)))  __bf16 v8bf;
typedef __attribute__((ext_vector_type(8)))  float  v8f;

__device__ __forceinline__ unsigned short f32_to_bf16(float f) {
  union { float f; unsigned u; } v; v.f = f;
  unsigned r = v.u + 0x7FFFu + ((v.u >> 16) & 1u);   // round-to-nearest-even
  return (unsigned short)(r >> 16);
}

// Async DMA: 16 bytes per lane, global -> LDS. Tracked by ASYNCcnt.
__device__ __forceinline__ void async_copy_b128(const unsigned short* gsrc,
                                                unsigned short* ldst) {
  // Flat LDS aperture keeps the LDS byte offset in addr[31:0].
  const unsigned lds = (unsigned)(unsigned long long)ldst;
  asm volatile("global_load_async_to_lds_b128 %0, %1, off"
               :: "v"(lds), "v"(gsrc) : "memory");
}

template <int N>
__device__ __forceinline__ void wait_asynccnt() {
  asm volatile("s_wait_asynccnt %0" :: "i"(N) : "memory");
}

// ---------------------------------------------------------------------------
// f32 -> bf16 elementwise convert
// ---------------------------------------------------------------------------
__global__ void cvt_f32_bf16(const float* __restrict__ in,
                             unsigned short* __restrict__ out, long long n) {
  long long i = (long long)blockIdx.x * blockDim.x + threadIdx.x;
  long long stride = (long long)gridDim.x * blockDim.x;
  for (; i < n; i += stride) out[i] = f32_to_bf16(in[i]);
}

// ---------------------------------------------------------------------------
// Generic batched WMMA GEMM: C[m][n] = sum_k A[m][k] * BT[n][k]  (+ epilogue)
//   A: bf16 row-major [M][K]   (per-batch stride batchStrideA, leading dim lda)
//   B: bf16 "BT" layout [N][K] (per-batch stride batchStrideB, leading dim ldb)
// Output index (MODE_QKV splits n into (h = n>>nShift, d = n&mask)):
//   oidx = (m >> rowsPerBShift)*strideB2 + (m & mask)*strideM
//        + batch*strideBatch + h*strideH + d*strideN
// NI = wave n-tiles (2 -> BN=64, 4 -> BN=128); wave grid 4(M) x 2(N); BM=128.
// Requires: M % 128 == 0, N % (32*NI) == 0, K % 64 == 0 (true for all uses).
// ---------------------------------------------------------------------------
enum { MODE_QKV = 0, MODE_SCORES = 1, MODE_AV = 2, MODE_FFN1 = 3, MODE_FFN2 = 4 };

struct GemmParams {
  const unsigned short* A; long long lda, batchStrideA;
  const unsigned short* B; long long ldb, batchStrideB;
  const float* bias;
  void* out;
  int rowsPerBShift;                                  // rowsPerB = 1<<shift
  int nShift;                                         // QKV: head split of n
  long long strideB2, strideM, strideBatch, strideN, strideH;
  const unsigned char* mask; long long maskLd;        // mask idx = n*maskLd + m
  float scale;
  long long K;
};

#define BM 128
#define BK 64
#define LDA_P 72   // padded LDS row (bf16 elems): 144B stride, conflict-free
#define LDB_P 72
#define NBUF 3     // triple-buffered LDS (DMA depth 2 + 1 being consumed)

template <int MODE, int NI>
__global__ __launch_bounds__(256, 2) void gemm_wmma(GemmParams p) {
  constexpr int BNt = 32 * NI;          // block-N extent
  constexpr int NBR = BNt / 32;         // B-tile rows per copy pass
  constexpr int CPI = 4 + NBR;          // async copies issued per thread/slab
  extern __shared__ unsigned short smem[];
  unsigned short* As = smem;                               // NBUF x BM  x LDA_P
  unsigned short* Bs = smem + NBUF * (BM * LDA_P);         // NBUF x BNt x LDB_P

  const int tid  = threadIdx.x;
  const int wave = tid >> 5;
  const int lane = tid & 31;
  const int wm   = wave & 3;   // 4 wave-rows of 32
  const int wn   = wave >> 2;  // 2 wave-cols of 16*NI
  const int g    = lane >> 4;  // lane half (WMMA layout)
  const int l16  = lane & 15;

  const int batch = blockIdx.z;
  const long long blockM = (long long)blockIdx.y * BM;
  const long long blockN = (long long)blockIdx.x * BNt;

  const unsigned short* Ab = p.A + (long long)batch * p.batchStrideA;
  const unsigned short* Bb = p.B + (long long)batch * p.batchStrideB;

  v8f c[2][NI] = {};

  const int lr    = tid & 7;   // which 8-elem chunk of the 64-wide K slab
  const int arow0 = tid >> 3;  // 0..31

  // issue an async DMA of one K-slab into LDS buffer `buf`
  auto issueSlab = [&](long long kb, int buf) {
#pragma unroll
    for (int r = 0; r < 4; ++r)
      async_copy_b128(Ab + (blockM + arow0 + r * 32) * p.lda + kb + lr * 8,
                      &As[buf * (BM * LDA_P) + (arow0 + r * 32) * LDA_P + lr * 8]);
#pragma unroll
    for (int r = 0; r < NBR; ++r)
      async_copy_b128(Bb + (blockN + arow0 + r * 32) * p.ldb + kb + lr * 8,
                      &Bs[buf * (BNt * LDB_P) + (arow0 + r * 32) * LDB_P + lr * 8]);
  };

  const int nslab = (int)(p.K >> 6);
  issueSlab(0, 0);
  if (nslab > 1) issueSlab(BK, 1);

  int bi = 0;  // LDS buffer holding slab i
  for (int i = 0; i < nslab; ++i) {
    // Retire slab i's DMA (async loads complete in order); keep slab i+1's
    // CPI copies in flight. Barrier makes every wave's slab-i DMA visible.
    if (i + 1 < nslab) wait_asynccnt<CPI>(); else wait_asynccnt<0>();
    __syncthreads();

    // Issue slab i+2 into the buffer last read at iteration i-1 (fenced by
    // the barrier above). DMA proceeds while we run WMMAs below.
    if (i + 2 < nslab) {
      int b2 = bi + 2; if (b2 >= NBUF) b2 -= NBUF;
      issueSlab((long long)(i + 2) * BK, b2);
      const long long kpf = (long long)(i + 3) * BK;
      if (kpf < p.K) {                        // warm L2 beyond the DMA window
        __builtin_prefetch(Ab + (blockM + arow0) * p.lda + kpf + lr * 8, 0, 1);
        __builtin_prefetch(Bb + (blockN + arow0) * p.ldb + kpf + lr * 8, 0, 1);
      }
    }

    const unsigned short* Abuf = &As[bi * (BM * LDA_P)];
    const unsigned short* Bbuf = &Bs[bi * (BNt * LDB_P)];

#pragma unroll
    for (int kk = 0; kk < 2; ++kk) {           // 2 WMMA K-steps of 32
      union U { v16bf v; v8bf h[2]; } a[2], b[NI];
#pragma unroll
      for (int mi = 0; mi < 2; ++mi) {
        // A 16x32 bf16 frag: lane half g gets K chunks {g*8, g*8+16}
        const unsigned short* src =
            &Abuf[(wm * 32 + mi * 16 + l16) * LDA_P + kk * 32 + g * 8];
        a[mi].h[0] = *(const v8bf*)(src);
        a[mi].h[1] = *(const v8bf*)(src + 16);
      }
#pragma unroll
      for (int ni = 0; ni < NI; ++ni) {
        // B 32x16 frag: lane n=l16, half g gets K chunks {g*8, g*8+16}
        const unsigned short* src =
            &Bbuf[(wn * (16 * NI) + ni * 16 + l16) * LDB_P + kk * 32 + g * 8];
        b[ni].h[0] = *(const v8bf*)(src);
        b[ni].h[1] = *(const v8bf*)(src + 16);
      }
#pragma unroll
      for (int mi = 0; mi < 2; ++mi)
#pragma unroll
        for (int ni = 0; ni < NI; ++ni)
          c[mi][ni] = __builtin_amdgcn_wmma_f32_16x16x32_bf16(
              false, a[mi].v, false, b[ni].v, (short)0, c[mi][ni], false, false);
    }
    ++bi; if (bi >= NBUF) bi = 0;
  }

  // ---- epilogue (all index math is shift/mask, no division) ----
  const long long mmMask = ((long long)1 << p.rowsPerBShift) - 1;
  const long long nMask  = ((long long)1 << p.nShift) - 1;
#pragma unroll
  for (int mi = 0; mi < 2; ++mi) {
    const long long m0  = blockM + wm * 32 + mi * 16;
    const long long bo  = m0 >> p.rowsPerBShift;
    const long long mm0 = m0 & mmMask;
#pragma unroll
    for (int ni = 0; ni < NI; ++ni) {
      const long long n = blockN + wn * (16 * NI) + ni * 16 + l16;
#pragma unroll
      for (int r = 0; r < 8; ++r) {
        const long long mm = mm0 + g * 8 + r;
        const long long m  = m0 + g * 8 + r;
        float v = c[mi][ni][r];
        if (MODE == MODE_QKV) {
          v += p.bias[n];
          const long long oidx = bo * p.strideB2 + mm * p.strideM +
                                 (n >> p.nShift) * p.strideH +
                                 (n & nMask) * p.strideN;
          ((unsigned short*)p.out)[oidx] = f32_to_bf16(v);
        } else {
          const long long oidx = bo * p.strideB2 + mm * p.strideM +
                                 (long long)batch * p.strideBatch + n * p.strideN;
          if (MODE == MODE_SCORES) {
            v *= p.scale;
            if (p.mask[n * p.maskLd + m]) v = -1000.0f;
            ((float*)p.out)[oidx] = v;
          } else if (MODE == MODE_AV) {
            ((float*)p.out)[oidx] = v;
          } else if (MODE == MODE_FFN1) {
            v += p.bias[n];
            v = v > 0.0f ? v : 0.0f;
            ((unsigned short*)p.out)[oidx] = f32_to_bf16(v);
          } else {  // MODE_FFN2
            v += p.bias[n];
            ((float*)p.out)[oidx] = v;
          }
        }
      }
    }
  }
}

// ---------------------------------------------------------------------------
// Softmax over the QUERY axis (reference softmaxes scores over dim s).
// Input: scT f32 [H][S(t)][S(s)]  -> softmax along the s (row) direction.
// Output: attn bf16 [H][S(s)][S(t)]   (transposed back for the attn*V GEMM).
// grid = (S/16, H), block = 256 (16 t-rows x 16 reducers).
// ---------------------------------------------------------------------------
__global__ __launch_bounds__(256) void softmax_query_axis(
    const float* __restrict__ scT, unsigned short* __restrict__ attn, int S) {
  const int h  = blockIdx.y;
  const int t0 = blockIdx.x * 16;
  const int tid = threadIdx.x;
  const int i = tid >> 4;   // row within tile
  const int j = tid & 15;   // reducer lane
  const float* row = scT + (long long)h * S * S + (long long)(t0 + i) * S;

  float mx = -3.4e38f;
  for (int s = j; s < S; s += 16) mx = fmaxf(mx, row[s]);
#pragma unroll
  for (int o = 8; o >= 1; o >>= 1) mx = fmaxf(mx, __shfl_xor(mx, o, 16));

  float sum = 0.0f;
  for (int s = j; s < S; s += 16) sum += __expf(row[s] - mx);
#pragma unroll
  for (int o = 8; o >= 1; o >>= 1) sum += __shfl_xor(sum, o, 16);

  __shared__ float smx[16], sinv[16];
  if (j == 0) { smx[i] = mx; sinv[i] = 1.0f / sum; }
  __syncthreads();

  // write transposed: thread handles (s = s0+i, t = t0+j) -> contiguous t writes
  const float* base = scT + (long long)h * S * S;
  unsigned short* ab = attn + (long long)h * S * S;
  const float m2 = smx[j], v2 = sinv[j];
  for (int s0 = 0; s0 < S; s0 += 16) {
    const int s = s0 + i, t = t0 + j;
    const float e = __expf(base[(long long)t * S + s] - m2) * v2;
    ab[(long long)s * S + t] = f32_to_bf16(e);
  }
}

// ---------------------------------------------------------------------------
// z = LayerNorm(x)*gamma + beta + x  per row of length D.
// WRITE_BF16=1 -> write bf16 only (feeds FFN1); 0 -> write f32 (final output).
// ---------------------------------------------------------------------------
template <int WRITE_BF16>
__global__ __launch_bounds__(256) void layernorm_residual(
    const float* __restrict__ in, const float* __restrict__ gamma,
    const float* __restrict__ beta, float* __restrict__ outF,
    unsigned short* __restrict__ outB, int D) {
  const long long row = blockIdx.x;
  const float* x = in + row * D;
  __shared__ float red[256];

  float s = 0.0f;
  for (int c = threadIdx.x; c < D; c += 256) s += x[c];
  red[threadIdx.x] = s; __syncthreads();
  for (int o = 128; o >= 1; o >>= 1) {
    if (threadIdx.x < o) red[threadIdx.x] += red[threadIdx.x + o];
    __syncthreads();
  }
  const float mu = red[0] / D; __syncthreads();

  float v = 0.0f;
  for (int c = threadIdx.x; c < D; c += 256) { float d = x[c] - mu; v += d * d; }
  red[threadIdx.x] = v; __syncthreads();
  for (int o = 128; o >= 1; o >>= 1) {
    if (threadIdx.x < o) red[threadIdx.x] += red[threadIdx.x + o];
    __syncthreads();
  }
  const float var = red[0] / D;
  const float inv = rsqrtf(var + 1e-5f);

  for (int c = threadIdx.x; c < D; c += 256) {
    const float z = (x[c] - mu) * inv * gamma[c] + beta[c] + x[c];
    if (WRITE_BF16) outB[row * D + c] = f32_to_bf16(z);
    else            outF[row * D + c] = z;
  }
}

// ---------------------------------------------------------------------------
// Host-side orchestration
// ---------------------------------------------------------------------------
extern "C" void kernel_launch(void* const* d_in, const int* in_sizes, int n_in,
                              void* d_out, int out_size, void* d_ws, size_t ws_size,
                              hipStream_t stream) {
  (void)in_sizes; (void)n_in; (void)out_size; (void)ws_size;
  const int Bn = 4, S = 2048, E = 768, Dm = 768, H = 12, dh = 64, F = 3072;

  const float*         x     = (const float*)d_in[0];
  const unsigned char* mask  = (const unsigned char*)d_in[1];  // jnp bool -> 1B
  const float*         Wq    = (const float*)d_in[2];
  const float*         bq    = (const float*)d_in[3];
  const float*         Wk    = (const float*)d_in[4];
  const float*         bk    = (const float*)d_in[5];
  const float*         Wv    = (const float*)d_in[6];
  const float*         bv    = (const float*)d_in[7];
  const float*         gamma = (const float*)d_in[8];
  const float*         beta  = (const float*)d_in[9];
  const float*         W1    = (const float*)d_in[10];
  const float*         b1    = (const float*)d_in[11];
  const float*         W2    = (const float*)d_in[12];
  const float*         b2    = (const float*)d_in[13];
  float* out = (float*)d_out;

  char* ws = (char*)d_ws;
  size_t off = 0;
  auto take = [&](size_t bytes) { size_t r = off; off = (off + bytes + 255) & ~(size_t)255; return r; };

  unsigned short* xb  = (unsigned short*)(ws + take((size_t)Bn * S * E * 2));
  unsigned short* Wqb = (unsigned short*)(ws + take((size_t)Dm * E * 2));
  unsigned short* Wkb = (unsigned short*)(ws + take((size_t)Dm * E * 2));
  unsigned short* Wvb = (unsigned short*)(ws + take((size_t)Dm * E * 2));
  unsigned short* W1b = (unsigned short*)(ws + take((size_t)F * Dm * 2));
  unsigned short* W2b = (unsigned short*)(ws + take((size_t)Dm * F * 2));
  unsigned short* qb  = (unsigned short*)(ws + take((size_t)Bn * H * S * dh * 2)); // [b,h,s,d]
  unsigned short* kb_ = (unsigned short*)(ws + take((size_t)Bn * H * S * dh * 2)); // [b,h,s,d]
  unsigned short* vTb = (unsigned short*)(ws + take((size_t)Bn * H * dh * S * 2)); // [b,h,d,s]
  size_t scT_off  = take((size_t)H * S * S * 4);   // f32 scores^T, per-b chunk
  size_t attn_off = take((size_t)H * S * S * 2);   // bf16 attn [h][s][t], per-b
  float* aout = (float*)(ws + take((size_t)Bn * S * Dm * 4));
  // aliases (lifetimes do not overlap):
  float*          scT   = (float*)(ws + scT_off);
  unsigned short* attn  = (unsigned short*)(ws + attn_off);
  unsigned short* ab    = (unsigned short*)(ws + attn_off);                 // post-LN1 bf16
  unsigned short* h1    = (unsigned short*)(ws + scT_off);                  // FFN hidden bf16
  float*          b2out = (float*)(ws + scT_off + (((size_t)Bn * S * F * 2 + 255) & ~(size_t)255));

  auto shBytes = [](int NI) {
    return (size_t)NBUF * ((size_t)BM * LDA_P + (size_t)32 * NI * LDB_P) * 2;
  };

  // ---- 0) f32 -> bf16 conversions ----
  cvt_f32_bf16<<<2048, 256, 0, stream>>>(x,  xb,  (long long)Bn * S * E);
  cvt_f32_bf16<<<1024, 256, 0, stream>>>(Wq, Wqb, (long long)Dm * E);
  cvt_f32_bf16<<<1024, 256, 0, stream>>>(Wk, Wkb, (long long)Dm * E);
  cvt_f32_bf16<<<1024, 256, 0, stream>>>(Wv, Wvb, (long long)Dm * E);
  cvt_f32_bf16<<<2048, 256, 0, stream>>>(W1, W1b, (long long)F * Dm);
  cvt_f32_bf16<<<2048, 256, 0, stream>>>(W2, W2b, (long long)Dm * F);

  // ---- 1) QKV projections: flat [B*S,E] x [D,E]^T GEMMs, head-split epilogue ----
  {
    GemmParams p{};
    p.A = xb; p.lda = E; p.batchStrideA = 0;
    p.ldb = E; p.batchStrideB = 0; p.strideBatch = 0;
    p.rowsPerBShift = 11;  // rows per batch-elem = S = 2048
    p.nShift = 6;          // n = (h<<6) | d
    p.K = E;
    dim3 grid(Dm / 128, (Bn * S) / BM, 1);
    // q -> [b,h,s,d]
    p.B = Wqb; p.bias = bq; p.out = qb;
    p.strideB2 = (long long)S * Dm; p.strideM = dh;
    p.strideH = (long long)S * dh; p.strideN = 1;
    gemm_wmma<MODE_QKV, 4><<<grid, 256, shBytes(4), stream>>>(p);
    // k -> [b,h,s,d]
    p.B = Wkb; p.bias = bk; p.out = kb_;
    gemm_wmma<MODE_QKV, 4><<<grid, 256, shBytes(4), stream>>>(p);
    // v -> [b,h,d,s]  (transposed so attn*V needs no transpose loads)
    p.B = Wvb; p.bias = bv; p.out = vTb;
    p.strideB2 = (long long)Dm * S; p.strideM = 1;
    p.strideH = (long long)dh * S; p.strideN = S;
    gemm_wmma<MODE_QKV, 4><<<grid, 256, shBytes(4), stream>>>(p);
  }

  // ---- 2) attention, chunked per batch element b ----
  const float scale = 0.02209708691f;  // 1/sqrt(2048) (reference scales by sqrt(S))
  for (int b = 0; b < Bn; ++b) {
    // scoresT[h][t][s] = (K Q^T)/sqrt(S), masked where mask[b,s,t]
    {
      GemmParams p{};
      p.A = kb_ + (size_t)b * H * S * dh; p.lda = dh; p.batchStrideA = (long long)S * dh;
      p.B = qb  + (size_t)b * H * S * dh; p.ldb = dh; p.batchStrideB = (long long)S * dh;
      p.out = scT; p.rowsPerBShift = 11; p.nShift = 0;
      p.strideB2 = 0; p.strideM = S;
      p.strideBatch = (long long)S * S; p.strideN = 1;
      p.mask = mask + (size_t)b * S * S; p.maskLd = S;
      p.scale = scale; p.K = dh;
      dim3 grid(S / 128, S / BM, H);
      gemm_wmma<MODE_SCORES, 4><<<grid, 256, shBytes(4), stream>>>(p);
    }
    // softmax over query axis; emits attn bf16 [h][s][t]
    {
      dim3 grid(S / 16, H);
      softmax_query_axis<<<grid, 256, 0, stream>>>(scT, attn, S);
    }
    // o[b,s,(h,d)] = attn * V   (concat heads directly into aout)
    {
      GemmParams p{};
      p.A = attn; p.lda = S; p.batchStrideA = (long long)S * S;
      p.B = vTb + (size_t)b * Dm * S; p.ldb = S; p.batchStrideB = (long long)dh * S;
      p.out = aout + (size_t)b * S * Dm;
      p.rowsPerBShift = 11; p.nShift = 0;
      p.strideB2 = 0; p.strideM = Dm;
      p.strideBatch = dh; p.strideN = 1; p.K = S;
      dim3 grid(1, S / BM, H);
      gemm_wmma<MODE_AV, 2><<<grid, 256, shBytes(2), stream>>>(p);
    }
  }

  // ---- 3) LN1 + residual -> bf16 activation for FFN ----
  layernorm_residual<1><<<Bn * S, 256, 0, stream>>>(aout, gamma, beta, nullptr, ab, Dm);

  // ---- 4) FFN1: relu(a W1^T + b1) -> bf16 ----
  {
    GemmParams p{};
    p.A = ab; p.lda = Dm; p.batchStrideA = 0;
    p.B = W1b; p.ldb = Dm; p.batchStrideB = 0;
    p.bias = b1; p.out = h1;
    p.rowsPerBShift = 13; p.nShift = 0;
    p.strideB2 = 0; p.strideM = F;
    p.strideBatch = 0; p.strideN = 1; p.K = Dm;
    dim3 grid(F / 128, (Bn * S) / BM, 1);
    gemm_wmma<MODE_FFN1, 4><<<grid, 256, shBytes(4), stream>>>(p);
  }

  // ---- 5) FFN2: h W2^T + b2 -> f32 ----
  {
    GemmParams p{};
    p.A = h1; p.lda = F; p.batchStrideA = 0;
    p.B = W2b; p.ldb = F; p.batchStrideB = 0;
    p.bias = b2; p.out = b2out;
    p.rowsPerBShift = 13; p.nShift = 0;
    p.strideB2 = 0; p.strideM = Dm;
    p.strideBatch = 0; p.strideN = 1; p.K = F;
    dim3 grid(Dm / 128, (Bn * S) / BM, 1);
    gemm_wmma<MODE_FFN2, 4><<<grid, 256, shBytes(4), stream>>>(p);
  }

  // ---- 6) LN2 + residual -> final f32 output ----
  layernorm_residual<0><<<Bn * S, 256, 0, stream>>>(b2out, gamma, beta, out, nullptr, Dm);
}
